// TPSelfAttention_74131135529308
// MI455X (gfx1250) — matrix operand order repaired and checked by reading, not compiled, not run on hardware
//
#include <hip/hip_runtime.h>
#include <hip/hip_bf16.h>

typedef unsigned short u16;
typedef __attribute__((ext_vector_type(16))) __bf16 bf16x16;
typedef __attribute__((ext_vector_type(8)))  float  floatx8;

union Frag {
    bf16x16 v;
    uint4   q[2];
    u16     s[16];
};

// 16-byte global -> LDS copy: async DMA (ASYNCcnt-tracked) on gfx1250,
// VGPR bounce otherwise. VDST = LDS byte address (low 32 bits of generic
// pointer, per aperture truncation rule), VADDR = 64-bit global address.
__device__ __forceinline__ void copy16_g2l(u16* dst_lds, const u16* src_g) {
#if defined(__gfx1250__)
    unsigned lds_addr = (unsigned)(size_t)dst_lds;
    unsigned long long gaddr = (unsigned long long)(size_t)src_g;
    asm volatile("global_load_async_to_lds_b128 %0, %1, off"
                 :: "v"(lds_addr), "v"(gaddr)
                 : "memory");
#else
    *(uint4*)dst_lds = *(const uint4*)src_g;
#endif
}

__device__ __forceinline__ void wait_async_zero() {
#if defined(__gfx1250__)
#if __has_builtin(__builtin_amdgcn_s_wait_asynccnt)
    __builtin_amdgcn_s_wait_asynccnt(0);
#else
    asm volatile("s_wait_asynccnt 0x0" ::: "memory");
#endif
#endif
}

__device__ __forceinline__ u16 f32_bf16(float f) {
    unsigned u = __float_as_uint(f);
    u += 0x7FFFu + ((u >> 16) & 1u);   // round-to-nearest-even
    return (u16)(u >> 16);
}

// ---------------------------------------------------------------------------
// fp32 -> bf16 conversion, 4 elements / thread
// ---------------------------------------------------------------------------
__global__ void k_cvt(const float* __restrict__ in, u16* __restrict__ out, int n4) {
    int i = blockIdx.x * blockDim.x + threadIdx.x;
    if (i < n4) {
        float4 f = ((const float4*)in)[i];
        u16 r[4] = { f32_bf16(f.x), f32_bf16(f.y), f32_bf16(f.z), f32_bf16(f.w) };
        ((uint2*)out)[i] = *(const uint2*)r;
    }
}

// ---------------------------------------------------------------------------
// C[M,N] = A[M,K] * W[N,K]^T   (bf16 in, fp32 accum)
// Block: 256 thr = 8 waves; tile 128(M) x 128(N), BK=32, double-buffered LDS.
// Waves as 2(M) x 4(N); wave tile 64x32 -> acc[4][2], 8 WMMA per K-step.
// Output: bf16 (Cb) or fp32+bias (Cf).
// ---------------------------------------------------------------------------
__global__ __launch_bounds__(256) void k_gemm(const u16* __restrict__ A,
                                              const u16* __restrict__ W,
                                              u16* __restrict__ Cb,
                                              float* __restrict__ Cf,
                                              const float* __restrict__ bias,
                                              int M, int N, int K) {
    __shared__ u16 As[2][128 * 32];
    __shared__ u16 Ws[2][128 * 32];

    const int tid  = threadIdx.x;
    const int lane = tid & 31;
    const int wave = tid >> 5;
    const int lh   = lane >> 4;   // half-wave select
    const int ll   = lane & 15;
    const int wm   = wave >> 2;   // 0..1
    const int wn   = wave & 3;    // 0..3
    const int bm   = blockIdx.y * 128;
    const int bn   = blockIdx.x * 128;

    // this thread's 2 A-chunks + 2 W-chunks (8 bf16 each) per K-step
    const int c0   = tid << 1;                 // chunks c0, c0+1 of 512
    const int row0 = c0 >> 2, cg0 = (c0 & 3) << 3;
    const int row1 = (c0 + 1) >> 2, cg1 = ((c0 + 1) & 3) << 3;

    floatx8 acc[4][2] = {};

    // prologue: stage k0 = 0 into buffer 0
    copy16_g2l(&As[0][row0 * 32 + cg0], &A[(size_t)(bm + row0) * K + cg0]);
    copy16_g2l(&As[0][row1 * 32 + cg1], &A[(size_t)(bm + row1) * K + cg1]);
    copy16_g2l(&Ws[0][row0 * 32 + cg0], &W[(size_t)(bn + row0) * K + cg0]);
    copy16_g2l(&Ws[0][row1 * 32 + cg1], &W[(size_t)(bn + row1) * K + cg1]);
    wait_async_zero();
    __syncthreads();

    int p = 0;
    for (int k0 = 0; k0 < K; k0 += 32) {
        // prefetch next K-step into the other buffer (overlaps WMMA below)
        if (k0 + 32 < K) {
            int kn = k0 + 32;
            copy16_g2l(&As[p ^ 1][row0 * 32 + cg0], &A[(size_t)(bm + row0) * K + kn + cg0]);
            copy16_g2l(&As[p ^ 1][row1 * 32 + cg1], &A[(size_t)(bm + row1) * K + kn + cg1]);
            copy16_g2l(&Ws[p ^ 1][row0 * 32 + cg0], &W[(size_t)(bn + row0) * K + kn + cg0]);
            copy16_g2l(&Ws[p ^ 1][row1 * 32 + cg1], &W[(size_t)(bn + row1) * K + kn + cg1]);
        }

        // fragments from current buffer
        Frag a[4], b[2];
#pragma unroll
        for (int ms = 0; ms < 4; ++ms) {
            int r = wm * 64 + ms * 16 + ll;
            a[ms].q[0] = *(const uint4*)&As[p][r * 32 + lh * 8];
            a[ms].q[1] = *(const uint4*)&As[p][r * 32 + 16 + lh * 8];
        }
#pragma unroll
        for (int ns = 0; ns < 2; ++ns) {
            int c = wn * 32 + ns * 16 + ll;
            b[ns].q[0] = *(const uint4*)&Ws[p][c * 32 + lh * 8];
            b[ns].q[1] = *(const uint4*)&Ws[p][c * 32 + 16 + lh * 8];
        }
#pragma unroll
        for (int ms = 0; ms < 4; ++ms)
#pragma unroll
            for (int ns = 0; ns < 2; ++ns)
                acc[ms][ns] = __builtin_amdgcn_wmma_f32_16x16x32_bf16(
                    false, a[ms].v, false, b[ns].v, (short)0, acc[ms][ns], false, false);

        wait_async_zero();   // next buffer's DMA done (after the math, not before)
        __syncthreads();
        p ^= 1;
    }

    // C layout: VGPR r holds row r (lanes 0-15) / row r+8 (lanes 16-31), col = ll
#pragma unroll
    for (int ms = 0; ms < 4; ++ms)
#pragma unroll
        for (int ns = 0; ns < 2; ++ns)
#pragma unroll
            for (int r = 0; r < 8; ++r) {
                int m = bm + wm * 64 + ms * 16 + r + 8 * lh;
                int n = bn + wn * 32 + ns * 16 + ll;
                float v = acc[ms][ns][r];
                if (Cb) Cb[(size_t)m * N + n] = f32_bf16(v);
                else    Cf[(size_t)m * N + n] = v + bias[n];
            }
}

// ---------------------------------------------------------------------------
// Flash-style causal attention.
// qkv: bf16 [B*T][3D], cols [0,2048)=Q [2048,4096)=K [4096,6144)=V
// out: bf16 [B*T][2048]
// Block = 128 thr (4 waves) handles (b, h, 64 q-rows); wave -> 16 q-rows.
// ---------------------------------------------------------------------------
__global__ __launch_bounds__(128) void k_attn(const u16* __restrict__ qkv,
                                              u16* __restrict__ out) {
    constexpr int T = 2048, D3 = 6144, D = 2048, HD = 128;
    const float scale = 0.08838834764831845f;  // 1/sqrt(128)

    __shared__ u16 Kls[32 * 128];      // 32 keys x 128 hd
    __shared__ u16 Vt[128 * 32];       // V transposed: hd x key
    __shared__ u16 Pls[4][16 * 32];    // per-wave P relayout buffer

    const int bid  = blockIdx.x;
    const int b    = bid >> 9;          // H * (T/64) = 512 blocks per batch
    const int rem  = bid & 511;
    const int h    = rem >> 5;
    const int q0   = (rem & 31) << 6;   // 64-row q tile
    const int tid  = threadIdx.x;
    const int lane = tid & 31;
    const int wave = tid >> 5;
    const int lh   = lane >> 4;
    const int ll   = lane & 15;

    // Q fragments for this wave's 16 rows: 4 chunks of K=32 over HD
    Frag qf[4];
    {
        int tq = q0 + wave * 16 + ll;
        const u16* qrow = qkv + (size_t)(b * T + tq) * D3 + h * HD;
#pragma unroll
        for (int kc = 0; kc < 4; ++kc) {
            qf[kc].q[0] = *(const uint4*)&qrow[kc * 32 + lh * 8];
            qf[kc].q[1] = *(const uint4*)&qrow[kc * 32 + 16 + lh * 8];
        }
    }

    float   m_run[8], l_run[8];
    floatx8 o[8] = {};
#pragma unroll
    for (int r = 0; r < 8; ++r) { m_run[r] = -3.0e38f; l_run[r] = 0.0f; }

    const int njb = (q0 + 64) >> 5;     // causal: key blocks of 32 up to tile end

    for (int j = 0; j < njb; ++j) {
        const int kbase = j << 5;
        // stage K via async DMA; stage V transposed via explicit scatter
#pragma unroll
        for (int i = 0; i < 4; ++i) {
            int c = tid + (i << 7);     // 512 chunks of 8 elems
            int row = c >> 4, cg = (c & 15) << 3;
            const u16* src = qkv + (size_t)(b * T + kbase + row) * D3 + h * HD;
            copy16_g2l(&Kls[row * 128 + cg], &src[D + cg]);
            uint4 vv = *(const uint4*)&src[2 * D + cg];
            const u16* vp = (const u16*)&vv;
#pragma unroll
            for (int e = 0; e < 8; ++e) Vt[(cg + e) * 32 + row] = vp[e];
        }
        wait_async_zero();
        __syncthreads();

        // S = Q K^T for 2 key sub-tiles of 16
        floatx8 sacc[2] = {};
#pragma unroll
        for (int ns = 0; ns < 2; ++ns) {
            int n = ns * 16 + ll;       // key index in block
#pragma unroll
            for (int kc = 0; kc < 4; ++kc) {
                Frag bk;
                bk.q[0] = *(const uint4*)&Kls[n * 128 + kc * 32 + lh * 8];
                bk.q[1] = *(const uint4*)&Kls[n * 128 + kc * 32 + 16 + lh * 8];
                sacc[ns] = __builtin_amdgcn_wmma_f32_16x16x32_bf16(
                    false, qf[kc].v, false, bk.v, (short)0, sacc[ns], false, false);
            }
        }

        // scale + causal mask
        const int trow0 = q0 + wave * 16 + 8 * lh;
        float sv[2][8];
#pragma unroll
        for (int ns = 0; ns < 2; ++ns)
#pragma unroll
            for (int r = 0; r < 8; ++r) {
                int kg = kbase + ns * 16 + ll;
                float v = sacc[ns][r] * scale;
                if (kg > trow0 + r) v += -1e9f;
                sv[ns][r] = v;
            }

        // online softmax: row max / exp / row sum (reduce over 16-lane group)
        float mnew[8], alpha[8], rsum[8];
#pragma unroll
        for (int r = 0; r < 8; ++r) {
            float mx = fmaxf(sv[0][r], sv[1][r]);
#pragma unroll
            for (int off = 8; off >= 1; off >>= 1)
                mx = fmaxf(mx, __shfl_xor(mx, off, 32));
            mnew[r]  = fmaxf(m_run[r], mx);
            alpha[r] = __expf(m_run[r] - mnew[r]);
            m_run[r] = mnew[r];
            rsum[r]  = 0.0f;
        }
#pragma unroll
        for (int ns = 0; ns < 2; ++ns)
#pragma unroll
            for (int r = 0; r < 8; ++r) {
                float pv = __expf(sv[ns][r] - mnew[r]);
                rsum[r] += pv;
                // C-layout element (row r+8*lh, col ns*16+ll) -> LDS row-major P
                Pls[wave][(r + 8 * lh) * 32 + ns * 16 + ll] = f32_bf16(pv);
            }
#pragma unroll
        for (int r = 0; r < 8; ++r) {
            float s = rsum[r];
#pragma unroll
            for (int off = 8; off >= 1; off >>= 1)
                s += __shfl_xor(s, off, 32);
            l_run[r] = alpha[r] * l_run[r] + s;
        }
#pragma unroll
        for (int f = 0; f < 8; ++f)
#pragma unroll
            for (int r = 0; r < 8; ++r) o[f][r] *= alpha[r];

        // P as A fragment (16x32), V^T as B fragments, accumulate O
        Frag pa;
        pa.q[0] = *(const uint4*)&Pls[wave][ll * 32 + lh * 8];
        pa.q[1] = *(const uint4*)&Pls[wave][ll * 32 + 16 + lh * 8];
#pragma unroll
        for (int ht = 0; ht < 8; ++ht) {
            int n = ht * 16 + ll;       // hd column
            Frag vb;
            vb.q[0] = *(const uint4*)&Vt[n * 32 + lh * 8];
            vb.q[1] = *(const uint4*)&Vt[n * 32 + 16 + lh * 8];
            o[ht] = __builtin_amdgcn_wmma_f32_16x16x32_bf16(
                false, pa.v, false, vb.v, (short)0, o[ht], false, false);
        }
        __syncthreads();
    }

    // normalize and store bf16 [B*T][D], head h columns
#pragma unroll
    for (int r = 0; r < 8; ++r) {
        float inv = 1.0f / l_run[r];
        int t = q0 + wave * 16 + r + 8 * lh;
        u16* orow = out + (size_t)(b * T + t) * D + h * HD;
#pragma unroll
        for (int ht = 0; ht < 8; ++ht)
            orow[ht * 16 + ll] = f32_bf16(o[ht][r] * inv);
    }
}

// ---------------------------------------------------------------------------
extern "C" void kernel_launch(void* const* d_in, const int* in_sizes, int n_in,
                              void* d_out, int out_size, void* d_ws, size_t ws_size,
                              hipStream_t stream) {
    (void)in_sizes; (void)n_in; (void)out_size; (void)ws_size;
    const float* x      = (const float*)d_in[0];
    // d_in[1] = causal_mask (recomputed inline; not read)
    const float* w_qkv  = (const float*)d_in[2];
    const float* w_out  = (const float*)d_in[3];
    const float* b_out  = (const float*)d_in[4];
    float* outp         = (float*)d_out;

    char* ws = (char*)d_ws;
    u16* xb    = (u16*)(ws + 0);            //  4096*2048 bf16 = 16 MB
    u16* wqkvb = (u16*)(ws + 16777216);     //  6144*2048 bf16 = 24 MB
    u16* woutb = (u16*)(ws + 41943040);     //  2048*2048 bf16 =  8 MB
    u16* qkvb  = (u16*)(ws + 50331648);     //  4096*6144 bf16 = 48 MB
    u16* attnb = (u16*)(ws + 100663296);    //  4096*2048 bf16 = 16 MB

    // fp32 -> bf16
    {
        int n4;
        n4 = (4096 * 2048) / 4;
        k_cvt<<<(n4 + 255) / 256, 256, 0, stream>>>(x, xb, n4);
        n4 = (6144 * 2048) / 4;
        k_cvt<<<(n4 + 255) / 256, 256, 0, stream>>>(w_qkv, wqkvb, n4);
        n4 = (2048 * 2048) / 4;
        k_cvt<<<(n4 + 255) / 256, 256, 0, stream>>>(w_out, woutb, n4);
    }

    // QKV projection: [4096,2048] x [6144,2048]^T -> bf16 [4096,6144]
    k_gemm<<<dim3(6144 / 128, 4096 / 128), 256, 0, stream>>>(
        xb, wqkvb, qkvb, nullptr, nullptr, 4096, 6144, 2048);

    // fused causal attention -> bf16 [4096,2048]
    k_attn<<<1024, 128, 0, stream>>>(qkvb, attnb);

    // output projection + bias: fp32 out
    k_gemm<<<dim3(2048 / 128, 4096 / 128), 256, 0, stream>>>(
        attnb, woutb, nullptr, outp, b_out, 4096, 2048, 2048);
}
